// SoftmaxContrastiveLoss_53609781789113
// MI455X (gfx1250) — matrix operand order repaired and checked by reading, not compiled, not run on hardware
//
#include <hip/hip_runtime.h>
#include <hip/hip_bf16.h>
#include <stdint.h>

#define N_ROWS 8192
#define DIM 256
#define NSPLIT 8
#define COLS_PER_SPLIT (N_ROWS / NSPLIT)      // 1024
#define TILES_PER_SPLIT (COLS_PER_SPLIT / 16) // 64
#define ROWS_PER_BLOCK 128
#define WAVES_PER_BLOCK 8
#define LDS_ROW 264                            // 256 bf16 + 8 bf16 (16B) TDM pad
#define EPS_DIST 1e-8f
#define EPS_CLAMP 1e-8f

typedef __attribute__((ext_vector_type(16))) __bf16 v16bf;
typedef __attribute__((ext_vector_type(8)))  __bf16 v8bf;
typedef __attribute__((ext_vector_type(8)))  float  v8f;
typedef __attribute__((ext_vector_type(4)))  float  v4f;
typedef __attribute__((ext_vector_type(4)))  unsigned int u32x4;
typedef __attribute__((ext_vector_type(8)))  int    i32x8;
typedef __attribute__((ext_vector_type(4)))  int    i32x4;

static __device__ __forceinline__ __bf16 f2bf(float f) {
    // round-to-nearest-even fp32 -> bf16
    unsigned u = __builtin_bit_cast(unsigned, f);
    u += 0x7FFFu + ((u >> 16) & 1u);
    unsigned short h = (unsigned short)(u >> 16);
    return __builtin_bit_cast(__bf16, h);
}

// ---------------------------------------------------------------------------
// TDM: stage 512 contiguous bf16 (two B-tile columns) from global into LDS,
// padding 16B after every 512B so staged rows have a 528B (conflict-free)
// stride. 1-D descriptor: tensor_dim0 = tile_dim0 = 512 elements, 2B each.
// ---------------------------------------------------------------------------
static __device__ __forceinline__ void tdm_stage_2rows(const __bf16* gsrc, __bf16* lds_dst) {
    unsigned long long ga = (unsigned long long)(uintptr_t)gsrc;
    unsigned la = (unsigned)(uintptr_t)lds_dst;   // low 32 bits = LDS offset
    int ga_lo = __builtin_amdgcn_readfirstlane((int)(unsigned)(ga & 0xFFFFFFFFull));
    int ga_hi = __builtin_amdgcn_readfirstlane((int)(unsigned)(ga >> 32));
    int la_s  = __builtin_amdgcn_readfirstlane((int)la);

    u32x4 g0;
    g0[0] = 1u;                                            // count=1, user D#
    g0[1] = (unsigned)la_s;                                // lds_addr (bytes)
    g0[2] = (unsigned)ga_lo;                               // global_addr[31:0]
    g0[3] = ((unsigned)ga_hi & 0x01FFFFFFu) | (2u << 30);  // addr[56:32] | type=2

    i32x8 g1;
    g1[0] = (int)0x07910000u;   // data_size=2B, pad_enable, interval=512B, amount=16B
    g1[1] = (int)(512u << 16);  // tensor_dim0 = 512 (bits 79:48, low half)
    g1[2] = (int)(1u << 16);    // tensor_dim1 = 1
    g1[3] = (int)(512u << 16);  // tile_dim0 = 512 (bits 127:112)
    g1[4] = 0;                  // tile_dim1 = tile_dim2 = 0 (unused -> 1D)
    g1[5] = 512;                // tensor_dim0_stride = 512
    g1[6] = 0;
    g1[7] = 0;

    i32x4 gz = {0, 0, 0, 0};
#if defined(__clang_major__) && (__clang_major__ >= 23)
    i32x8 gz8 = {0, 0, 0, 0, 0, 0, 0, 0};
    __builtin_amdgcn_tensor_load_to_lds(g0, g1, gz, gz, gz8, 0);
#else
    __builtin_amdgcn_tensor_load_to_lds(g0, g1, gz, gz, 0);
#endif
}

// ---------------------------------------------------------------------------
// Kernel 1: L2-normalize rows of f and a, write bf16 copies to workspace.
// One wave32 per row; 8 floats per lane; shuffle reduction for the norm.
// ---------------------------------------------------------------------------
__global__ void __launch_bounds__(256)
normalize_bf16_kernel(const float* __restrict__ f, const float* __restrict__ a,
                      __bf16* __restrict__ fb, __bf16* __restrict__ ab) {
    const int wave = threadIdx.x >> 5;
    const int lane = threadIdx.x & 31;
    const int gw = blockIdx.x * (blockDim.x >> 5) + wave; // 0..16383
    const int row = gw & (N_ROWS - 1);
    const float* src = (gw < N_ROWS) ? f : a;
    __bf16* dst = (gw < N_ROWS) ? fb : ab;

    const v4f* p = (const v4f*)(src + (size_t)row * DIM + lane * 8);
    v4f x0 = p[0];
    v4f x1 = p[1];
    float ss = 0.f;
#pragma unroll
    for (int i = 0; i < 4; ++i) { ss += x0[i] * x0[i]; ss += x1[i] * x1[i]; }
#pragma unroll
    for (int m = 16; m >= 1; m >>= 1) ss += __shfl_xor(ss, m, 32);

    float scale = 1.0f / fmaxf(__builtin_amdgcn_sqrtf(ss), 1e-12f);
    v8bf o;
#pragma unroll
    for (int i = 0; i < 4; ++i) {
        o[i]     = f2bf(x0[i] * scale);
        o[i + 4] = f2bf(x1[i] * scale);
    }
    *(v8bf*)(dst + (size_t)row * DIM + lane * 8) = o;
}

// ---------------------------------------------------------------------------
// Kernel 2: fused bf16-WMMA Gram matrix + online row softmax partials.
// Block = 8 waves = 128-row block; all waves share one column split.
// B tiles (16 cols x 256 K, contiguous 8KB) are TDM-staged into LDS,
// double-buffered: DMA of tile t+1 overlaps WMMA+softmax of tile t.
// A strip (16 x 256 bf16) lives entirely in VGPRs per wave; all 8 B
// fragments are preloaded per tile so the 8 WMMAs issue back-to-back.
// ---------------------------------------------------------------------------
__global__ void __launch_bounds__(256)
gemm_softmax_partial_kernel(const __bf16* __restrict__ fb,
                            const __bf16* __restrict__ ab,
                            float* __restrict__ part) {
    __shared__ __bf16 sbuf[2][16 * LDS_ROW];   // 2 x 8448 B, padded rows

    const int lane = threadIdx.x & 31;
    const int wave = threadIdx.x >> 5;
    const int half = lane >> 4;   // 0: rows 0-7 / K-lo group, 1: rows 8-15 / K-hi
    const int nlo  = lane & 15;
    const int rowbase = blockIdx.x * ROWS_PER_BLOCK + wave * 16;
    const int split   = blockIdx.y;
    const int colbase = split * COLS_PER_SPLIT;

    // Preload all A fragments for this wave's 16-row strip (64 VGPRs).
    // 16-bit A 16x32 layout: lane holds row M=nlo, K = kb + half*8 + {0..7}
    // in VGPRs 0-3 and K = kb + half*8 + 16 + {0..7} in VGPRs 4-7.
    v16bf afrag[8];
    {
        const __bf16* arow = fb + (size_t)(rowbase + nlo) * DIM;
#pragma unroll
        for (int kk = 0; kk < 8; ++kk) {
            v8bf lo = *(const v8bf*)(arow + kk * 32 + half * 8);
            v8bf hi = *(const v8bf*)(arow + kk * 32 + half * 8 + 16);
            afrag[kk] = __builtin_shufflevector(lo, hi,
                0, 1, 2, 3, 4, 5, 6, 7, 8, 9, 10, 11, 12, 13, 14, 15);
        }
    }

    // Online softmax running state: per lane, 8 rows x (max, sum).
    float m[8], s[8];
#pragma unroll
    for (int r = 0; r < 8; ++r) { m[r] = -1e30f; s[r] = 0.f; }

    // Prologue: every wave TDM-stages its 2-column slice of tile 0.
    tdm_stage_2rows(ab + (size_t)(colbase + wave * 2) * DIM,
                    &sbuf[0][wave * 2 * LDS_ROW]);

    for (int t = 0; t < TILES_PER_SPLIT; ++t) {
        __builtin_amdgcn_s_wait_tensorcnt(0);  // own slice of tile t landed
        __syncthreads();                       // whole tile t visible; buf[(t+1)&1] free

        if (t + 1 < TILES_PER_SPLIT) {         // overlap DMA of t+1 with compute of t
            tdm_stage_2rows(ab + (size_t)(colbase + (t + 1) * 16 + wave * 2) * DIM,
                            &sbuf[(t + 1) & 1][wave * 2 * LDS_ROW]);
        }

        // 16-bit B 32x16 layout: lane holds column N=nlo,
        // K = kb + half*8 + {0..7} and kb + half*8 + 16 + {0..7}, matching A.
        // Preload all 8 fragments (16x ds_load_b128 as one clause), then
        // run the 8 WMMAs back-to-back to keep the XDL pipe full.
        const __bf16* bcol = &sbuf[t & 1][nlo * LDS_ROW];
        v16bf bfrag[8];
#pragma unroll
        for (int kk = 0; kk < 8; ++kk) {
            v8bf lo = *(const v8bf*)(bcol + kk * 32 + half * 8);
            v8bf hi = *(const v8bf*)(bcol + kk * 32 + half * 8 + 16);
            bfrag[kk] = __builtin_shufflevector(lo, hi,
                0, 1, 2, 3, 4, 5, 6, 7, 8, 9, 10, 11, 12, 13, 14, 15);
        }

        v8f acc = {};
#pragma unroll
        for (int kk = 0; kk < 8; ++kk) {
            acc = __builtin_amdgcn_wmma_f32_16x16x32_bf16(
                false, afrag[kk], false, bfrag[kk], (short)0, acc, false, false);
        }

        // logits + online max/sum update for the 8 rows this lane holds
#pragma unroll
        for (int r = 0; r < 8; ++r) {
            float sim   = acc[r];
            float d2    = fmaxf(2.0f - 2.0f * sim, 0.0f);
            float dist  = __builtin_amdgcn_sqrtf(d2);
            float logit = fmaxf(__builtin_amdgcn_rcpf(dist + EPS_DIST), EPS_CLAMP);
            float mn = fmaxf(m[r], logit);
            s[r] = s[r] * __expf(m[r] - mn) + __expf(logit - mn);
            m[r] = mn;
        }
    }

    // Merge the 16 column-lanes within each half-wave (xor masks stay in-half).
#pragma unroll
    for (int mask = 1; mask <= 8; mask <<= 1) {
#pragma unroll
        for (int r = 0; r < 8; ++r) {
            float mo = __shfl_xor(m[r], mask, 32);
            float so = __shfl_xor(s[r], mask, 32);
            float mn = fmaxf(m[r], mo);
            s[r] = s[r] * __expf(m[r] - mn) + so * __expf(mo - mn);
            m[r] = mn;
        }
    }

    if (nlo == 0) {
#pragma unroll
        for (int r = 0; r < 8; ++r) {
            int row = rowbase + half * 8 + r;
            float* p = part + (size_t)(row * NSPLIT + split) * 2;
            p[0] = m[r];
            p[1] = s[r];
        }
    }
}

// ---------------------------------------------------------------------------
// Kernel 3: deterministic finalize. Merge the NSPLIT partials per row,
// recompute the diagonal logit, fixed-order tree reduction to the scalar.
// ---------------------------------------------------------------------------
__global__ void __launch_bounds__(1024)
finalize_kernel(const __bf16* __restrict__ fb, const __bf16* __restrict__ ab,
                const float* __restrict__ part, float* __restrict__ out) {
    __shared__ float red[1024];
    const int tid = threadIdx.x;
    float acc = 0.f;

    for (int i = tid; i < N_ROWS; i += 1024) {
        float m = -1e30f, s = 0.f;
#pragma unroll
        for (int sp = 0; sp < NSPLIT; ++sp) {
            const float* p = part + (size_t)(i * NSPLIT + sp) * 2;
            float mi = p[0], si = p[1];
            float mn = fmaxf(m, mi);
            s = s * __expf(m - mn) + si * __expf(mi - mn);
            m = mn;
        }
        // diagonal logit from the bf16 normalized rows
        float dot = 0.f;
        const __bf16* fr = fb + (size_t)i * DIM;
        const __bf16* ar = ab + (size_t)i * DIM;
        for (int k = 0; k < DIM; k += 8) {
            v8bf x = *(const v8bf*)(fr + k);
            v8bf y = *(const v8bf*)(ar + k);
#pragma unroll
            for (int j = 0; j < 8; ++j) dot += (float)x[j] * (float)y[j];
        }
        float d2   = fmaxf(2.0f - 2.0f * dot, 0.0f);
        float dlog = fmaxf(__builtin_amdgcn_rcpf(__builtin_amdgcn_sqrtf(d2) + EPS_DIST),
                           EPS_CLAMP);
        // -logp[i,i] = m + log(s) - logit_diag
        acc += m + __logf(s) - dlog;
    }

    red[tid] = acc;
    __syncthreads();
    for (int off = 512; off > 0; off >>= 1) {
        if (tid < off) red[tid] += red[tid + off];
        __syncthreads();
    }
    if (tid == 0) out[0] = red[0] / (float)N_ROWS;
}

// ---------------------------------------------------------------------------
extern "C" void kernel_launch(void* const* d_in, const int* in_sizes, int n_in,
                              void* d_out, int out_size, void* d_ws, size_t ws_size,
                              hipStream_t stream) {
    (void)in_sizes; (void)n_in; (void)out_size; (void)ws_size;
    const float* f = (const float*)d_in[0];
    const float* a = (const float*)d_in[1];
    float* out = (float*)d_out;

    char* ws = (char*)d_ws;
    __bf16* fb  = (__bf16*)ws;                                    // 4 MiB
    __bf16* ab  = (__bf16*)(ws + (size_t)N_ROWS * DIM * 2);       // 4 MiB
    float* part = (float*)(ws + (size_t)N_ROWS * DIM * 4);        // 512 KiB

    // 16384 rows total (f then a), 8 waves (256 threads) per block
    normalize_bf16_kernel<<<2048, 256, 0, stream>>>(f, a, fb, ab);

    // 64 row-blocks x 8 column splits, 8 waves per block sharing TDM-staged B
    gemm_softmax_partial_kernel<<<dim3(N_ROWS / ROWS_PER_BLOCK, NSPLIT), 256, 0, stream>>>(
        fb, ab, part);

    finalize_kernel<<<1, 1024, 0, stream>>>(fb, ab, part, out);
}
